// CosineAttention_84576495993322
// MI455X (gfx1250) — compile-verified, matrix operand order checked by reading, not run on hardware
//
#include <hip/hip_runtime.h>
#include <hip/hip_bf16.h>
#include <math.h>

typedef __attribute__((ext_vector_type(16))) _Float16 v16h;
typedef __attribute__((ext_vector_type(8)))  float    v8f;
typedef __attribute__((ext_vector_type(4)))  int      v4i;

typedef __attribute__((address_space(1))) v4i* global_v4i_ptr;
typedef __attribute__((address_space(3))) v4i* lds_v4i_ptr;

#define WMMA_F16(a, b, c) \
  __builtin_amdgcn_wmma_f32_16x16x32_f16(false, (a), false, (b), (short)0, (c), false, false)

#if __has_builtin(__builtin_amdgcn_global_load_async_to_lds_b128)
#define ATTN_ASYNC 1
#else
#define ATTN_ASYNC 0
#endif

// ---------------------------------------------------------------------------
// Fragment loaders per CDNA5 16-bit A/B VGPR layouts (ISA 7.12.2):
//   A 16x32 (MxK): lane%16 = M row; lanes 0-15 hold K=0..7 / K=16..23
//                  (elems 0-7 / 8-15); lanes 16-31 hold K=8..15 / K=24..31.
//   B 32x16 (KxN): lane%16 = N col; same K striping per lane half.
//   C/D 16x16 f32: vgpr v -> row v (lanes 0-15) / v+8 (lanes 16-31), col lane%16.
// Sources are "row-major with a full K strip per row"; each 8-element run is
// 16 contiguous bytes -> one b128 (global or ds) load per run.
// ---------------------------------------------------------------------------
__device__ __forceinline__ v16h frag_ld_h(const _Float16* __restrict__ base, int ld, int k0) {
  const int l  = threadIdx.x & 31;
  const int r  = l & 15;
  const int kb = (l < 16) ? 0 : 8;
  const _Float16* p = base + (size_t)r * ld + k0 + kb;
  v16h f;
#pragma unroll
  for (int e = 0; e < 8; ++e) f[e]     = p[e];
#pragma unroll
  for (int e = 0; e < 8; ++e) f[e + 8] = p[e + 16];
  return f;
}

// ---------------------------------------------------------------------------
// Cooperative 64x64 f16 tile stage: global (row-major, leading dim ld) -> LDS
// (leading dim 64). 128 threads x 4 chunks of 16B. Uses the gfx1250 async
// global->LDS DMA path (ASYNCcnt) when available.
// ---------------------------------------------------------------------------
__device__ __forceinline__ void stage_tile_64x64(const _Float16* g, int ld, _Float16* l) {
  const int t = threadIdx.x;  // 0..127
#pragma unroll
  for (int i = 0; i < 4; ++i) {
    const int q   = i * 128 + t;   // 0..511 16B chunks
    const int row = q >> 3;
    const int seg = (q & 7) * 8;
#if ATTN_ASYNC
    __builtin_amdgcn_global_load_async_to_lds_b128(
        (global_v4i_ptr)(void*)(g + (size_t)row * ld + seg),
        (lds_v4i_ptr)(void*)(l + row * 64 + seg), 0, 0);
#else
    *(uint4*)(l + row * 64 + seg) = *(const uint4*)(g + (size_t)row * ld + seg);
#endif
  }
}

__device__ __forceinline__ void wait_stage_and_barrier() {
#if ATTN_ASYNC
#if __has_builtin(__builtin_amdgcn_s_wait_asynccnt)
  __builtin_amdgcn_s_wait_asynccnt(0);
#else
  asm volatile("s_wait_asynccnt 0x0" ::: "memory");
#endif
#endif
  __syncthreads();
}

// ---------------------------------------------------------------------------
// Elementwise fp32 -> f16 convert (n multiple of 4).
// ---------------------------------------------------------------------------
__global__ void __launch_bounds__(256)
cvt_f16(const float* __restrict__ s, _Float16* __restrict__ d, int n) {
  int i = (int)(blockIdx.x * blockDim.x + threadIdx.x) * 4;
  if (i >= n) return;
#pragma unroll
  for (int e = 0; e < 4; ++e) d[i + e] = (_Float16)s[i + e];
}

// ---------------------------------------------------------------------------
// Per-batch transpose + convert: P (B x S x 1024 fp32) -> Vt (B x 1024 x S f16),
// Vt_b[c][s] = P_b[s][c]. LDS-tiled so both global streams stay coalesced.
// ---------------------------------------------------------------------------
__global__ void __launch_bounds__(256)
transpose_cvt_f16(const float* __restrict__ src, _Float16* __restrict__ dst) {
  __shared__ float t[32][33];
  const int b  = blockIdx.z;
  const float* S = src + (size_t)b * 2048 * 1024;
  _Float16*    D = dst + (size_t)b * 1024 * 2048;
  const int s0 = blockIdx.x * 32;
  const int c0 = blockIdx.y * 32;
  const int tx = threadIdx.x & 31;
  const int ty = threadIdx.x >> 5;
#pragma unroll
  for (int r = ty; r < 32; r += 8) t[r][tx] = S[(size_t)(s0 + r) * 1024 + c0 + tx];
  __syncthreads();
#pragma unroll
  for (int r = ty; r < 32; r += 8)
    D[(size_t)(c0 + r) * 2048 + s0 + tx] = (_Float16)t[tx][r];
}

// ---------------------------------------------------------------------------
// GEMM: C[4096,1024] = X @ W^T (+ bias), f16 in, f32 out. Block = 8 waves;
// wave tile 32x64 (8 accumulators); grid (16,16).
// ---------------------------------------------------------------------------
__global__ void __launch_bounds__(256)
gemm_h_xwT(const _Float16* __restrict__ X, const _Float16* __restrict__ W,
           const float* __restrict__ bias, float* __restrict__ C) {
  const int wave = threadIdx.x >> 5;
  const int row0 = blockIdx.x * 256 + wave * 32;
  const int col0 = blockIdx.y * 64;

  v8f acc[2][4];
#pragma unroll
  for (int r = 0; r < 2; ++r)
#pragma unroll
    for (int j = 0; j < 4; ++j) acc[r][j] = v8f{};

  for (int k0 = 0; k0 < 1024; k0 += 32) {
    const v16h a0 = frag_ld_h(X + (size_t)row0 * 1024,        1024, k0);
    const v16h a1 = frag_ld_h(X + (size_t)(row0 + 16) * 1024, 1024, k0);
#pragma unroll
    for (int j = 0; j < 4; ++j) {
      const v16h bw = frag_ld_h(W + (size_t)(col0 + j * 16) * 1024, 1024, k0);
      acc[0][j] = WMMA_F16(a0, bw, acc[0][j]);
      acc[1][j] = WMMA_F16(a1, bw, acc[1][j]);
    }
  }

  const int l  = threadIdx.x & 31;
  const int n  = l & 15;
  const int mo = (l < 16) ? 0 : 8;
  float* Cr = C + (size_t)row0 * 1024 + col0;
#pragma unroll
  for (int j = 0; j < 4; ++j) {
    const float bj = bias ? bias[col0 + j * 16 + n] : 0.0f;
#pragma unroll
    for (int r = 0; r < 2; ++r)
#pragma unroll
      for (int v = 0; v < 8; ++v)
        Cr[(size_t)(r * 16 + v + mo) * 1024 + j * 16 + n] = acc[r][j][v] + bj;
  }
}

// ---------------------------------------------------------------------------
// Per-head L2 normalize (64 contiguous floats) + fp32 -> f16. Wave per chunk.
// ---------------------------------------------------------------------------
__global__ void __launch_bounds__(256)
normalize_h(const float* __restrict__ src, _Float16* __restrict__ dst, int nchunks) {
  const int wid = (int)((blockIdx.x * blockDim.x + threadIdx.x) >> 5);
  const int l   = threadIdx.x & 31;
  if (wid >= nchunks) return;
  const float* p = src + (size_t)wid * 64;
  const float x0 = p[l * 2];
  const float x1 = p[l * 2 + 1];
  float ss = x0 * x0 + x1 * x1;
#pragma unroll
  for (int off = 16; off > 0; off >>= 1) ss += __shfl_xor(ss, off, 32);
  const float inv = 1.0f / fmaxf(sqrtf(ss), 1e-12f);
  _Float16* q = dst + (size_t)wid * 64;
  q[l * 2]     = (_Float16)(x0 * inv);
  q[l * 2 + 1] = (_Float16)(x1 * inv);
}

// ---------------------------------------------------------------------------
// Cosine attention. Scores in [-1,1] => exp() safe without running max.
//   Pass 1: stream K tiles (async-staged to LDS once per block, double
//           buffered), accumulate per-row sum(exp(score)).
//   Pass 2: re-stage K+V tiles, recompute scores, scale, single attn write,
//           stage P f16 in LDS (C-layout -> A-layout), ctx += P @ V.
// K/V tiles are shared by all 4 waves -> block-level staging removes the 4x
// redundant global loads. Block = 128 thr = 4 waves x 16 query rows;
// grid (32 qtiles, 32 b*h). EXEC stays full around every WMMA.
// ---------------------------------------------------------------------------
__global__ void __launch_bounds__(128)
cosine_attn_kernel(const _Float16* __restrict__ Qn, const _Float16* __restrict__ Kn,
                   const _Float16* __restrict__ Vt, float* __restrict__ attn,
                   float* __restrict__ ctx) {
  __shared__ _Float16 kbuf[2][64 * 64];  // keys x dims, double buffered (8 KB each)
  __shared__ _Float16 vbuf[2][64 * 64];  // dims x keys, double buffered
  __shared__ _Float16 pl[4][16 * 64];    // per-wave probability staging

  const int wave = threadIdx.x >> 5;
  const int bh   = blockIdx.y;          // b*16 + h
  const int b    = bh >> 4;
  const int h    = bh & 15;
  const int q0   = blockIdx.x * 64 + wave * 16;

  const _Float16* Qb = Qn + ((size_t)b * 2048 + q0) * 1024 + h * 64;
  const _Float16* Kb = Kn + (size_t)b * 2048 * 1024 + h * 64;   // rows = keys, ld 1024
  const _Float16* Vb = Vt + (size_t)bh * 64 * 2048;             // rows = dims, ld 2048

  const v16h aq0 = frag_ld_h(Qb, 1024, 0);
  const v16h aq1 = frag_ld_h(Qb, 1024, 32);

  // ---- pass 1: softmax denominators -------------------------------------
  float se[8];
#pragma unroll
  for (int v = 0; v < 8; ++v) se[v] = 0.0f;

  stage_tile_64x64(Kb, 1024, kbuf[0]);
  wait_stage_and_barrier();

  for (int kt = 0; kt < 2048; kt += 64) {
    const int bi = (kt >> 6) & 1;
    if (kt + 64 < 2048)
      stage_tile_64x64(Kb + (size_t)(kt + 64) * 1024, 1024, kbuf[bi ^ 1]);
    const _Float16* kb = kbuf[bi];
#pragma unroll
    for (int j = 0; j < 4; ++j) {
      const v16h b0 = frag_ld_h(kb + (size_t)(j * 16) * 64, 64, 0);
      const v16h b1 = frag_ld_h(kb + (size_t)(j * 16) * 64, 64, 32);
      v8f s = {};
      s = WMMA_F16(aq0, b0, s);
      s = WMMA_F16(aq1, b1, s);
#pragma unroll
      for (int v = 0; v < 8; ++v) se[v] += __expf(s[v]);
    }
    wait_stage_and_barrier();
  }
#pragma unroll
  for (int v = 0; v < 8; ++v) {
#pragma unroll
    for (int off = 1; off < 16; off <<= 1) se[v] += __shfl_xor(se[v], off, 32);
  }
  float rinv[8];
#pragma unroll
  for (int v = 0; v < 8; ++v) rinv[v] = 1.0f / se[v];

  // ---- pass 2: write attn once, accumulate ctx --------------------------
  const int l  = threadIdx.x & 31;
  const int n  = l & 15;
  const int mo = (l < 16) ? 0 : 8;
  float* attnR = attn + ((size_t)bh * 2048 + q0) * 2048;
  _Float16* plw = pl[wave];

  v8f cacc[4] = {v8f{}, v8f{}, v8f{}, v8f{}};

  stage_tile_64x64(Kb, 1024, kbuf[0]);
  stage_tile_64x64(Vb, 2048, vbuf[0]);
  wait_stage_and_barrier();

  for (int kt = 0; kt < 2048; kt += 64) {
    const int bi = (kt >> 6) & 1;
    if (kt + 64 < 2048) {
      stage_tile_64x64(Kb + (size_t)(kt + 64) * 1024, 1024, kbuf[bi ^ 1]);
      stage_tile_64x64(Vb + kt + 64,                  2048, vbuf[bi ^ 1]);
    }
    const _Float16* kb = kbuf[bi];
    const _Float16* vb = vbuf[bi];
#pragma unroll
    for (int j = 0; j < 4; ++j) {
      const v16h b0 = frag_ld_h(kb + (size_t)(j * 16) * 64, 64, 0);
      const v16h b1 = frag_ld_h(kb + (size_t)(j * 16) * 64, 64, 32);
      v8f s = {};
      s = WMMA_F16(aq0, b0, s);
      s = WMMA_F16(aq1, b1, s);
#pragma unroll
      for (int v = 0; v < 8; ++v) {
        const float p = __expf(s[v]) * rinv[v];
        attnR[(size_t)(v + mo) * 2048 + kt + j * 16 + n] = p;   // the only attn write
        plw[(v + mo) * 64 + j * 16 + n] = (_Float16)p;          // stage for ctx GEMM
      }
    }
    // ctx += P(16q x 64keys) @ V(64keys x 64dims); A and B both from LDS.
#pragma unroll
    for (int c = 0; c < 2; ++c) {
      const v16h ap = frag_ld_h(plw, 64, c * 32);
#pragma unroll
      for (int j = 0; j < 4; ++j) {
        const v16h bv = frag_ld_h(vb + (size_t)(j * 16) * 64, 64, c * 32);
        cacc[j] = WMMA_F16(ap, bv, cacc[j]);
      }
    }
    wait_stage_and_barrier();
  }

  float* ctxR = ctx + ((size_t)b * 2048 + q0) * 1024 + h * 64;
#pragma unroll
  for (int j = 0; j < 4; ++j)
#pragma unroll
    for (int v = 0; v < 8; ++v)
      ctxR[(size_t)(v + mo) * 1024 + j * 16 + n] = cacc[j][v];
}

// ---------------------------------------------------------------------------
extern "C" void kernel_launch(void* const* d_in, const int* in_sizes, int n_in,
                              void* d_out, int out_size, void* d_ws, size_t ws_size,
                              hipStream_t stream) {
  const float* query = (const float*)d_in[0];
  const float* key   = (const float*)d_in[1];
  const float* value = (const float*)d_in[2];
  const float* w_q   = (const float*)d_in[3];
  const float* w_k   = (const float*)d_in[4];
  const float* w_v   = (const float*)d_in[5];
  const float* w_o   = (const float*)d_in[6];
  const float* b_o   = (const float*)d_in[7];

  float* out  = (float*)d_out;                       // (2,2048,1024)
  float* attn = out + (size_t)2 * 2048 * 1024;       // (2,16,2048,2048)

  char* ws = (char*)d_ws;
  float*    P  = (float*)ws;                                   // 16 MB fp32 GEMM out / ctx
  _Float16* Qn = (_Float16*)(ws + (size_t)16 * 1024 * 1024);   // 8 MB
  _Float16* Kn = Qn + (size_t)4096 * 1024;                     // 8 MB
  _Float16* Vt = Kn + (size_t)4096 * 1024;                     // 8 MB (transposed V)
  _Float16* Xh = Vt + (size_t)2 * 1024 * 2048;                 // 8 MB f16 activations
  _Float16* Wh = Xh + (size_t)4096 * 1024;                     // 2 MB f16 weights

  const int nX = 4096 * 1024;
  const int nW = 1024 * 1024;
  const dim3 ggrid(16, 16);     // 4096/256 x 1024/64
  const dim3 tgrid(64, 32, 2);  // 2048/32 x 1024/32 x B

  // ---- Q ----------------------------------------------------------------
  cvt_f16<<<nX / 1024, 256, 0, stream>>>(query, Xh, nX);
  cvt_f16<<<nW / 1024, 256, 0, stream>>>(w_q, Wh, nW);
  gemm_h_xwT<<<ggrid, 256, 0, stream>>>(Xh, Wh, nullptr, P);
  normalize_h<<<8192, 256, 0, stream>>>(P, Qn, 4096 * 16);
  // ---- K ----------------------------------------------------------------
  cvt_f16<<<nX / 1024, 256, 0, stream>>>(key, Xh, nX);
  cvt_f16<<<nW / 1024, 256, 0, stream>>>(w_k, Wh, nW);
  gemm_h_xwT<<<ggrid, 256, 0, stream>>>(Xh, Wh, nullptr, P);
  normalize_h<<<8192, 256, 0, stream>>>(P, Kn, 4096 * 16);
  // ---- V (transpose per (b,h) so ctx-B fragments are contiguous) --------
  cvt_f16<<<nX / 1024, 256, 0, stream>>>(value, Xh, nX);
  cvt_f16<<<nW / 1024, 256, 0, stream>>>(w_v, Wh, nW);
  gemm_h_xwT<<<ggrid, 256, 0, stream>>>(Xh, Wh, nullptr, P);
  transpose_cvt_f16<<<tgrid, 256, 0, stream>>>(P, Vt);

  // ---- attention: single attn write, ctx into P -------------------------
  cosine_attn_kernel<<<dim3(32, 32), 128, 0, stream>>>(Qn, Kn, Vt, attn, P);

  // ---- out = ctx @ w_o^T + b_o ------------------------------------------
  cvt_f16<<<nX / 1024, 256, 0, stream>>>(P, Xh, nX);
  cvt_f16<<<nW / 1024, 256, 0, stream>>>(w_o, Wh, nW);
  gemm_h_xwT<<<ggrid, 256, 0, stream>>>(Xh, Wh, b_o, out);
}
